// ParallelSelfAttention_19146964206370
// MI455X (gfx1250) — matrix operand order repaired and checked by reading, compiled, not run on hardware
//
#include <hip/hip_runtime.h>
#include <hip/hip_bf16.h>

// ---------------------------------------------------------------------------
// GPT self-attention block on MI455X (gfx1250, wave32, WMMA + TDM).
//   S=2048, B=1, H=2048, NH=16, HD=128
// Pipeline:
//   0) f32 -> bf16 pre-pass for hidden, w_qkv, w_dense (one-time, halves all
//      re-read traffic; removes cvt from GEMM hot loops)
//   1) QKV GEMM (WMMA bf16) + bias + rotary -> Q,K [NH][S][HD], V^T [NH][HD][S]
//   2) flash attention (WMMA bf16, online softmax, scores never hit HBM)
//   3) output projection (WMMA bf16) -> f32
// GEMM B-tiles are staged into LDS by the Tensor Data Mover
// (tensor_load_to_lds, TENSORcnt), issued by wave 0 of each block.
// Workspace (bytes): Q 8M | K 8M | Vt 8M | Ctx 8M | wq_bf 24M | wd_bf 8M |
// hid_bf 8M  = 72 MB.
// ---------------------------------------------------------------------------

#define SLEN 2048
#define HDIM 2048
#define NHEAD 16
#define HEADD 128

typedef __attribute__((ext_vector_type(16))) __bf16 v16bf;
typedef __attribute__((ext_vector_type(8)))  float  v8f;
typedef __attribute__((ext_vector_type(4)))  unsigned int u32x4;
typedef __attribute__((ext_vector_type(8)))  int    i32x8;
typedef __attribute__((ext_vector_type(4)))  int    i32x4;

// WMMA fragment maps (wave32):
//   A (16x32 bf16): lane L -> row L%16 ; halves 0..15 -> K = (L/16)*16 + i (contiguous)
//   B (32x16 bf16): lane L -> col L%16 ; half i -> K = (L/16)*16 + i (contiguous)
//   C/D (16x16 f32): VGPR r, lane L -> row = r + 8*(L/16), col = L%16

__device__ __forceinline__ v16bf load_bf16_frag32(const __bf16* base, int kb8) {
    // base -> 32 consecutive bf16 (this fragment row's K=0..31); pull the two
    // 8-half groups this lane owns (16B each, 16B aligned).
    const uint4* p = (const uint4*)base;
    v16bf f;
    ((uint4*)&f)[0] = p[kb8 >> 2];
    ((uint4*)&f)[1] = p[(kb8 >> 2) + 1];
    return f;
}

// ---------------------------------------------------------------------------
// TDM: DMA a 2D tile (tile_d0 x tile_d1 bf16, row stride stride_d0 elements)
// from global memory into LDS at byte offset lds_off (contiguous).
// D# layout per CDNA5 ISA ch.8: group0 = {flags, lds_addr, global_addr, type},
// group1 = {data_size, tensor dims, tile dims, dim0 stride}.
// ---------------------------------------------------------------------------
__device__ __forceinline__ void tdm_load_tile_2d(unsigned int lds_off,
                                                 const void* gsrc,
                                                 unsigned int tensor_d0,
                                                 unsigned int tensor_d1,
                                                 unsigned int tile_d0,
                                                 unsigned int tile_d1,
                                                 unsigned int stride_d0) {
    unsigned long long ga = (unsigned long long)(size_t)gsrc;
    u32x4 g0;
    g0[0] = 1u;                                            // count=1, user desc
    g0[1] = lds_off;                                       // lds_addr (bytes)
    g0[2] = (unsigned int)ga;                              // global_addr[31:0]
    g0[3] = (unsigned int)((ga >> 32) & 0x1FFFFFFull)      // global_addr[56:32]
          | (2u << 30);                                    // type = 2 (image)
    i32x8 g1;
    g1[0] = (int)(1u << 16);                               // data_size=1 (2 B)
    g1[1] = (int)((tensor_d0 & 0xFFFFu) << 16);            // tensor_dim0[15:0]
    g1[2] = (int)((tensor_d0 >> 16) & 0xFFFFu)             // tensor_dim0[31:16]
          | (int)((tensor_d1 & 0xFFFFu) << 16);            // tensor_dim1[15:0]
    g1[3] = (int)((tensor_d1 >> 16) & 0xFFFFu)             // tensor_dim1[31:16]
          | (int)(tile_d0 << 16);                          // tile_dim0
    g1[4] = (int)(tile_d1 & 0xFFFFu);                      // tile_dim1 (dim2=0)
    g1[5] = (int)stride_d0;                                // dim0_stride[31:0]
    g1[6] = 0;                                             // stride hi, dim1_stride lo
    g1[7] = 0;
    i32x4 z4 = {0, 0, 0, 0};
#if __clang_major__ >= 23
    i32x8 z8 = {};
    __builtin_amdgcn_tensor_load_to_lds(g0, g1, z4, z4, z8, 0);
#else
    __builtin_amdgcn_tensor_load_to_lds(g0, g1, z4, z4, 0);
#endif
}

// ---------------------------------------------------------------------------
// Kernel 0: elementwise f32 -> bf16 (8 elements / thread)
// ---------------------------------------------------------------------------
__global__ __launch_bounds__(256)
void cvt_bf16_kernel(const float* __restrict__ src, __bf16* __restrict__ dst,
                     unsigned int n8) {
    unsigned int i = blockIdx.x * 256u + threadIdx.x;
    if (i >= n8) return;
    size_t base = (size_t)i * 8;
    float4 a = *(const float4*)(src + base);
    float4 b = *(const float4*)(src + base + 4);
    __bf16 tmp[8] = {(__bf16)a.x, (__bf16)a.y, (__bf16)a.z, (__bf16)a.w,
                     (__bf16)b.x, (__bf16)b.y, (__bf16)b.z, (__bf16)b.w};
    *(uint4*)(dst + base) = *(const uint4*)tmp;
}

// ---------------------------------------------------------------------------
// Kernel 1: fused QKV GEMM + bias + rotary.  grid=(48,16), block=256 (8 waves)
//   blockIdx.x = segment (head*3 + {0:Q,1:K,2:V}) -> 128 output cols
//   blockIdx.y = 128-row block; wave w owns rows [base + 16w, +16)
// ---------------------------------------------------------------------------
__global__ __launch_bounds__(256)
void qkv_rope_kernel(const __bf16* __restrict__ hid_bf,
                     const __bf16* __restrict__ wq_bf,
                     const float* __restrict__ b_qkv,
                     __bf16* __restrict__ Qb,   // [NH][S][HD]
                     __bf16* __restrict__ Kb,   // [NH][S][HD]
                     __bf16* __restrict__ Vt)   // [NH][HD][S]
{
    __shared__ __bf16 ldsB[128][32];           // 8 KB B tile, filled by TDM

    const int tid  = threadIdx.x;
    const int wave = tid >> 5;
    const int lane = tid & 31;
    const int l15  = lane & 15;
    const int lg   = lane >> 4;
    const int kb8  = lg * 8;

    const int seg  = blockIdx.x;
    const int head = seg / 3;
    const int part = seg % 3;                  // 0=Q 1=K 2=V
    const int o0   = head * 384 + part * 128;
    const int rowbase = blockIdx.y * 128 + wave * 16;

    const unsigned int lds_off = (unsigned int)(size_t)&ldsB[0][0]; // flat low32 = LDS offset

    v8f zero = {};
    v8f acc[8];
#pragma unroll
    for (int t = 0; t < 8; ++t) acc[t] = zero;

    for (int k0 = 0; k0 < HDIM; k0 += 32) {
        __syncthreads();                       // prior-iter LDS reads done
        if (wave == 0) {                       // TDM: stage 128x32 bf16 B tile
            tdm_load_tile_2d(lds_off, wq_bf + (size_t)o0 * HDIM + k0,
                             HDIM, 3 * HDIM, 32, 128, HDIM);
            __builtin_amdgcn_s_wait_tensorcnt((short)0);
        }
        __syncthreads();

        const __bf16* ap = hid_bf + (size_t)(rowbase + l15) * HDIM + k0;
        __builtin_prefetch(ap + 32, 0, 1);
        v16bf afrag = load_bf16_frag32(ap, kb8);

#pragma unroll
        for (int t = 0; t < 8; ++t) {
            v16bf bfrag = *(const v16bf*)&ldsB[t * 16 + l15][lg * 16];
            acc[t] = __builtin_amdgcn_wmma_f32_16x16x32_bf16(
                false, afrag, false, bfrag, (short)0, acc[t], false, false);
        }
    }

    // bias
#pragma unroll
    for (int t = 0; t < 8; ++t) {
        float bv = b_qkv[o0 + t * 16 + l15];
#pragma unroll
        for (int r = 0; r < 8; ++r) acc[t][r] += bv;
    }

    // rotary: pair (d, d+64) lives in tiles (t, t+4), same lane -> register-only
    if (part < 2) {
        const float nlog2b = -0.20762050593046016f;  // -log2(10000)/64
#pragma unroll
        for (int r = 0; r < 8; ++r) {
            float srow = (float)(rowbase + r + 8 * lg);
#pragma unroll
            for (int t = 0; t < 4; ++t) {
                int dlo = t * 16 + l15;
                float theta = srow * exp2f((float)dlo * nlog2b);
                float sn, cs;
                __sincosf(theta, &sn, &cs);
                float lo = acc[t][r], hi = acc[t + 4][r];
                acc[t][r]     = lo * cs - hi * sn;
                acc[t + 4][r] = hi * cs + lo * sn;
            }
        }
    }

#pragma unroll
    for (int t = 0; t < 8; ++t) {
#pragma unroll
        for (int r = 0; r < 8; ++r) {
            int srow = rowbase + r + 8 * lg;
            int d    = t * 16 + l15;
            __bf16 v = (__bf16)acc[t][r];
            if (part == 0)      Qb[((size_t)head * SLEN + srow) * HEADD + d] = v;
            else if (part == 1) Kb[((size_t)head * SLEN + srow) * HEADD + d] = v;
            else                Vt[((size_t)head * HEADD + d) * SLEN + srow] = v;
        }
    }
}

// ---------------------------------------------------------------------------
// Kernel 2: flash attention.  grid=(S/64, NH), block=128 (4 waves).
// Each wave owns 16 query rows; online softmax over 32-key blocks.
// ---------------------------------------------------------------------------
__global__ __launch_bounds__(128)
void attn_kernel(const __bf16* __restrict__ Qb,
                 const __bf16* __restrict__ Kb,
                 const __bf16* __restrict__ Vt,
                 __bf16* __restrict__ Cb)      // [S][H]
{
    __shared__ __bf16 ldsP[4][16 * 32];        // per-wave P re-layout slab

    const int tid  = threadIdx.x;
    const int wave = tid >> 5;
    const int lane = tid & 31;
    const int l15  = lane & 15;
    const int lg   = lane >> 4;
    const int kb8  = lg * 8;

    const int head = blockIdx.y;
    const int q0   = blockIdx.x * 64 + wave * 16;

    const __bf16* Qh = Qb + (size_t)head * SLEN * HEADD;
    const __bf16* Kh = Kb + (size_t)head * SLEN * HEADD;
    const __bf16* Vh = Vt + (size_t)head * HEADD * SLEN;

    v16bf qf[4];
    {
        const __bf16* qp = Qh + (size_t)(q0 + l15) * HEADD;
#pragma unroll
        for (int c = 0; c < 4; ++c) qf[c] = load_bf16_frag32(qp + c * 32, kb8);
    }

    v8f zero = {};
    v8f acc[8];
#pragma unroll
    for (int t = 0; t < 8; ++t) acc[t] = zero;
    float m[8], l[8];
#pragma unroll
    for (int r = 0; r < 8; ++r) { m[r] = -1e30f; l[r] = 0.0f; }

    const float scale = 0.08838834764831845f;  // 1/sqrt(128)
    const int nblk = (q0 + 47) >> 5;           // causal: keys <= q0+15

    for (int jb = 0; jb < nblk; ++jb) {
        const int j0 = jb * 32;

        v8f sc[2];
#pragma unroll
        for (int nt = 0; nt < 2; ++nt) {
            v8f s = zero;
            const __bf16* kp = Kh + (size_t)(j0 + nt * 16 + l15) * HEADD + lg * 16;
#pragma unroll
            for (int c = 0; c < 4; ++c) {
                v16bf bfrag = *(const v16bf*)(kp + c * 32);
                s = __builtin_amdgcn_wmma_f32_16x16x32_bf16(
                    false, qf[c], false, bfrag, (short)0, s, false, false);
            }
            sc[nt] = s;
        }

        // scale + causal mask (reference fill = -10000)
#pragma unroll
        for (int nt = 0; nt < 2; ++nt) {
#pragma unroll
            for (int r = 0; r < 8; ++r) {
                int col = j0 + nt * 16 + l15;
                int row = q0 + r + 8 * lg;
                float v = sc[nt][r] * scale;
                sc[nt][r] = (col > row) ? -10000.0f : v;
            }
        }

        // online softmax (row lives across 16 lanes of one group)
        float alpha[8];
#pragma unroll
        for (int r = 0; r < 8; ++r) {
            float mx = fmaxf(sc[0][r], sc[1][r]);
#pragma unroll
            for (int msk = 1; msk < 16; msk <<= 1)
                mx = fmaxf(mx, __shfl_xor(mx, msk, 32));
            float mnew = fmaxf(m[r], mx);
            alpha[r] = __expf(m[r] - mnew);
            float p0 = __expf(sc[0][r] - mnew);
            float p1 = __expf(sc[1][r] - mnew);
            sc[0][r] = p0; sc[1][r] = p1;
            float rs = p0 + p1;
#pragma unroll
            for (int msk = 1; msk < 16; msk <<= 1)
                rs += __shfl_xor(rs, msk, 32);
            l[r] = l[r] * alpha[r] + rs;
            m[r] = mnew;
        }
#pragma unroll
        for (int t = 0; t < 8; ++t)
#pragma unroll
            for (int r = 0; r < 8; ++r) acc[t][r] *= alpha[r];

        // C-layout P -> A-fragment layout via private LDS slab
        __bf16* pl = &ldsP[wave][0];
#pragma unroll
        for (int nt = 0; nt < 2; ++nt)
#pragma unroll
            for (int r = 0; r < 8; ++r)
                pl[(r + 8 * lg) * 32 + nt * 16 + l15] = (__bf16)sc[nt][r];
        asm volatile("s_wait_dscnt 0" ::: "memory");   // same-wave RAW in LDS
        v16bf pf = load_bf16_frag32(pl + l15 * 32, kb8);

        // PV: 8 d-tiles, V transposed so B-fragments are contiguous
#pragma unroll
        for (int dt = 0; dt < 8; ++dt) {
            const __bf16* vp = Vh + (size_t)(dt * 16 + l15) * SLEN + j0 + lg * 16;
            v16bf vfrag = *(const v16bf*)vp;
            acc[dt] = __builtin_amdgcn_wmma_f32_16x16x32_bf16(
                false, pf, false, vfrag, (short)0, acc[dt], false, false);
        }
    }

#pragma unroll
    for (int r = 0; r < 8; ++r) {
        float inv = 1.0f / l[r];
        int row = q0 + r + 8 * lg;
#pragma unroll
        for (int dt = 0; dt < 8; ++dt) {
            int h = head * HEADD + dt * 16 + l15;
            Cb[(size_t)row * HDIM + h] = (__bf16)(acc[dt][r] * inv);
        }
    }
}

// ---------------------------------------------------------------------------
// Kernel 3: output projection  out = Ctx @ w_dense^T  (bias skipped:
// reference uses skip_bias_add).  grid=(16,16), block=256 (8 waves).
// ---------------------------------------------------------------------------
__global__ __launch_bounds__(256)
void dense_kernel(const __bf16* __restrict__ Cb,
                  const __bf16* __restrict__ wd_bf,
                  float* __restrict__ out)
{
    __shared__ __bf16 ldsB[128][32];

    const int tid  = threadIdx.x;
    const int wave = tid >> 5;
    const int lane = tid & 31;
    const int l15  = lane & 15;
    const int lg   = lane >> 4;
    const int kb8  = lg * 8;

    const int o0      = blockIdx.x * 128;
    const int rowbase = blockIdx.y * 128 + wave * 16;

    const unsigned int lds_off = (unsigned int)(size_t)&ldsB[0][0];

    v8f zero = {};
    v8f acc[8];
#pragma unroll
    for (int t = 0; t < 8; ++t) acc[t] = zero;

    for (int k0 = 0; k0 < HDIM; k0 += 32) {
        __syncthreads();
        if (wave == 0) {
            tdm_load_tile_2d(lds_off, wd_bf + (size_t)o0 * HDIM + k0,
                             HDIM, HDIM, 32, 128, HDIM);
            __builtin_amdgcn_s_wait_tensorcnt((short)0);
        }
        __syncthreads();

        const __bf16* ap = Cb + (size_t)(rowbase + l15) * HDIM + k0;
        __builtin_prefetch(ap + 32, 0, 1);
        v16bf afrag = load_bf16_frag32(ap, kb8);

#pragma unroll
        for (int t = 0; t < 8; ++t) {
            v16bf bfrag = *(const v16bf*)&ldsB[t * 16 + l15][lg * 16];
            acc[t] = __builtin_amdgcn_wmma_f32_16x16x32_bf16(
                false, afrag, false, bfrag, (short)0, acc[t], false, false);
        }
    }

#pragma unroll
    for (int t = 0; t < 8; ++t)
#pragma unroll
        for (int r = 0; r < 8; ++r) {
            int row = rowbase + r + 8 * lg;
            out[(size_t)row * HDIM + o0 + t * 16 + l15] = acc[t][r];
        }
}

// ---------------------------------------------------------------------------
extern "C" void kernel_launch(void* const* d_in, const int* in_sizes, int n_in,
                              void* d_out, int out_size, void* d_ws, size_t ws_size,
                              hipStream_t stream) {
    const float* hidden  = (const float*)d_in[0];
    // d_in[1] = attention_mask (bool causal triu) -> applied analytically
    const float* w_qkv   = (const float*)d_in[2];
    const float* b_qkv   = (const float*)d_in[3];
    const float* w_dense = (const float*)d_in[4];
    float*       out     = (float*)d_out;

    char* ws = (char*)d_ws;
    const size_t MB8 = (size_t)8 * 1024 * 1024;
    __bf16* Qb     = (__bf16*)(ws);                 //  8 MB
    __bf16* Kb     = (__bf16*)(ws + 1 * MB8);       //  8 MB
    __bf16* Vt     = (__bf16*)(ws + 2 * MB8);       //  8 MB
    __bf16* Cb     = (__bf16*)(ws + 3 * MB8);       //  8 MB
    __bf16* wq_bf  = (__bf16*)(ws + 4 * MB8);       // 24 MB
    __bf16* wd_bf  = (__bf16*)(ws + 7 * MB8);       //  8 MB
    __bf16* hid_bf = (__bf16*)(ws + 8 * MB8);       //  8 MB  (total 72 MB)

    // 0) one-time f32 -> bf16 down-converts
    {
        unsigned int n8;
        n8 = (3u * HDIM * HDIM) / 8;
        cvt_bf16_kernel<<<(n8 + 255) / 256, 256, 0, stream>>>(w_qkv, wq_bf, n8);
        n8 = (HDIM * HDIM) / 8;
        cvt_bf16_kernel<<<(n8 + 255) / 256, 256, 0, stream>>>(w_dense, wd_bf, n8);
        n8 = (SLEN * HDIM) / 8;
        cvt_bf16_kernel<<<(n8 + 255) / 256, 256, 0, stream>>>(hidden, hid_bf, n8);
    }

    qkv_rope_kernel<<<dim3(NHEAD * 3, SLEN / 128), 256, 0, stream>>>(
        hid_bf, wq_bf, b_qkv, Qb, Kb, Vt);
    attn_kernel<<<dim3(SLEN / 64, NHEAD), 128, 0, stream>>>(Qb, Kb, Vt, Cb);
    dense_kernel<<<dim3(HDIM / 128, SLEN / 128), 256, 0, stream>>>(
        Cb, wd_bf, out);
}
